// Encoder_21242908246505
// MI455X (gfx1250) — matrix operand order repaired and checked by reading, not compiled, not run on hardware
//
#include <hip/hip_runtime.h>
#include <hip/hip_bf16.h>
#include <math.h>

// ---------------------------------------------------------------------------
// Transformer encoder for MI455X (gfx1250): bf16 WMMA GEMMs + flash attention,
// with Tensor-Data-Mover (TDM) tile staging when the toolchain exposes it.
// L=4, H=16, DM=512, S=512, MU=2048, D=1024, B=8.
// ---------------------------------------------------------------------------

typedef __bf16 bf16;
typedef __attribute__((ext_vector_type(16))) __bf16 v16bf;
typedef __attribute__((ext_vector_type(8)))  float  v8f;
typedef unsigned __attribute__((ext_vector_type(4))) v4u;
typedef int      __attribute__((ext_vector_type(4))) v4i;
typedef int      __attribute__((ext_vector_type(8))) v8i;

#define L_LAYERS 4
#define H_HEADS  16
#define DMIN     512
#define SEQ      512
#define MUDIM    2048
#define DMODEL   1024
#define DH       64
#define BATCH    8
#define NROWS    (BATCH * SEQ)     /* 4096 */
#define LN_EPS   1e-6f

#if defined(__has_builtin)
#if __has_builtin(__builtin_amdgcn_tensor_load_to_lds) && \
    __has_builtin(__builtin_amdgcn_s_wait_tensorcnt)
#define CDNA5_HAS_TDM 1
#endif
#endif

union Frag16 { v16bf v; uint4 u[2]; };

__device__ __forceinline__ v8f v8f_zero() {
  v8f z;
#pragma unroll
  for (int i = 0; i < 8; ++i) z[i] = 0.0f;
  return z;
}

__device__ __forceinline__ v8f wmma_bf16(v16bf a, v16bf b, v8f c) {
  // D = A(16x32 bf16) * B(32x16 bf16) + C(16x16 f32)
  return __builtin_amdgcn_wmma_f32_16x16x32_bf16(
      /*neg_a=*/false, a, /*neg_b=*/false, b,
      /*c_mod=*/(short)0, c, /*reuse_a=*/false, /*reuse_b=*/false);
}

#ifdef CDNA5_HAS_TDM
// Issue one TDM 2D tile load: global (row-major, stride0 elems of 2B) -> LDS,
// with LDS row padding (pad codes per D# group1 spec). Wave-level op.
__device__ __forceinline__ void tdm_load_2d(
    unsigned lds_addr, unsigned long long gaddr,
    unsigned tensor_dim0, unsigned tensor_dim1, unsigned stride0,
    unsigned tile_cols, unsigned tile_rows,
    unsigned pad_iv, unsigned pad_amt) {
  v4u g0;
  g0.x = 1u;                                            // count=1 (valid D#)
  g0.y = lds_addr;                                      // LDS byte address
  g0.z = (unsigned)(gaddr & 0xffffffffull);             // global addr [31:0]
  g0.w = (unsigned)((gaddr >> 32) & 0x01ffffffull) |    // global addr [56:32]
         (2u << 30);                                    // D# type = 2 (image)
  v8i g1;
  g1[0] = (int)((1u << 16) |                            // data_size = 2 bytes
                (1u << 20) |                            // pad_enable
                (pad_iv << 22) | (pad_amt << 25));      // LDS row padding
  g1[1] = (int)((tensor_dim0 & 0xffffu) << 16);         // tensor_dim0[15:0]
  g1[2] = (int)(((tensor_dim0 >> 16) & 0xffffu) |
                ((tensor_dim1 & 0xffffu) << 16));       // dim0[31:16]|dim1[15:0]
  g1[3] = (int)(((tensor_dim1 >> 16) & 0xffffu) |
                ((tile_cols & 0xffffu) << 16));         // dim1[31:16]|tile_dim0
  g1[4] = (int)(tile_rows & 0xffffu);                   // tile_dim1 (dim2 = 0)
  g1[5] = (int)stride0;                                 // tensor_dim0_stride
  g1[6] = 0;
  g1[7] = 0;
  v4i g2 = {0, 0, 0, 0};
  v4i g3 = {0, 0, 0, 0};
#if defined(__clang_major__) && __clang_major__ >= 23
  v8i g4 = {0, 0, 0, 0, 0, 0, 0, 0};
  __builtin_amdgcn_tensor_load_to_lds(g0, g1, g2, g3, g4, 0);
#else
  __builtin_amdgcn_tensor_load_to_lds(g0, g1, g2, g3, 0);
#endif
}
#endif  // CDNA5_HAS_TDM

// ---------------------------------------------------------------------------
// h = concat(x, one_hot(pos))  -> f32 residual stream + bf16 compute copy
// ---------------------------------------------------------------------------
__global__ __launch_bounds__(256) void concat_kernel(
    const float* __restrict__ x, float* __restrict__ hf, bf16* __restrict__ hb) {
  int row = blockIdx.x;                  // b*SEQ + s
  int s   = row & (SEQ - 1);
  const float* xr = x + (size_t)row * DMIN;
  float* hfr = hf + (size_t)row * DMODEL;
  bf16*  hbr = hb + (size_t)row * DMODEL;
  for (int j = threadIdx.x; j < DMODEL; j += 256) {
    float v = (j < DMIN) ? xr[j] : ((j - DMIN) == s ? 1.0f : 0.0f);
    hfr[j] = v;
    hbr[j] = (bf16)v;
  }
}

// ---------------------------------------------------------------------------
// Weight prep (once per layer): W f32 [K][N]  ->  Wt bf16 [N][K]
// ---------------------------------------------------------------------------
__global__ __launch_bounds__(256) void wtrans_kernel(
    const float* __restrict__ W, bf16* __restrict__ Wt, int K, int N) {
  __shared__ float Ts[32][33];
  const int k0 = blockIdx.x * 32;
  const int n0 = blockIdx.y * 32;
  const int t  = threadIdx.x;
#pragma unroll
  for (int i = 0; i < 4; ++i) {
    int idx = t + i * 256;
    int r = idx >> 5, c = idx & 31;            // r: k, c: n
    Ts[r][c] = W[(size_t)(k0 + r) * N + n0 + c];
  }
  __syncthreads();
#pragma unroll
  for (int i = 0; i < 4; ++i) {
    int idx = t + i * 256;
    int r = idx >> 5, c = idx & 31;            // r: n, c: k
    Wt[(size_t)(n0 + r) * K + k0 + c] = (bf16)Ts[c][r];
  }
}

// ---------------------------------------------------------------------------
// V prep (per layer): vb bf16 [B,S,H*dh] -> Vt bf16 [B,H,dh,S]
// ---------------------------------------------------------------------------
__global__ __launch_bounds__(256) void vtrans_kernel(
    const bf16* __restrict__ V, bf16* __restrict__ Vt) {
  __shared__ bf16 Ts[32][40];
  const int bh = blockIdx.x;
  const int b  = bh >> 4, h = bh & 15;
  const int s0 = blockIdx.y * 32;
  const int d0 = blockIdx.z * 32;
  const int t  = threadIdx.x;
#pragma unroll
  for (int i = 0; i < 4; ++i) {
    int idx = t + i * 256;
    int r = idx >> 5, c = idx & 31;            // r: s, c: d
    Ts[r][c] = V[((size_t)(b * SEQ + s0 + r)) * DMODEL + h * DH + d0 + c];
  }
  __syncthreads();
#pragma unroll
  for (int i = 0; i < 4; ++i) {
    int idx = t + i * 256;
    int r = idx >> 5, c = idx & 31;            // r: d, c: s
    Vt[((size_t)(b * H_HEADS + h) * DH + d0 + r) * SEQ + s0 + c] = Ts[c][r];
  }
}

// ---------------------------------------------------------------------------
// GEMM: C[M,N] = A[M,K](bf16 row-major) x Bt[N,K](bf16, pre-transposed weights)
//       + bias, opt ReLU. Double-buffered LDS, one barrier per K-step.
// Block tile 128x128, 8 waves, wave tile 32x64 (2x4 WMMA accs), K-step 32.
// TDM stages tiles (wave 0 issues 2 descriptors/K-step); fallback = vector ld/st.
// ---------------------------------------------------------------------------
template <bool HASF, bool HASB, bool RELU>
__global__ __launch_bounds__(256) void gemm_kernel(
    const bf16* __restrict__ A, const bf16* __restrict__ Bt,
    const float* __restrict__ bias, float* __restrict__ outF,
    bf16* __restrict__ outB, int M, int N, int K) {
  (void)M;
  __shared__ bf16 As[2][128][40];   // [buf][m][k], +8 pad
  __shared__ bf16 Bs[2][128][40];   // [buf][n][k], +8 pad

  const int tid  = threadIdx.x;
  const int m0   = blockIdx.x * 128;
  const int n0   = blockIdx.y * 128;
  const int lane = tid & 31;
  const int wv   = tid >> 5;
  const int wr   = wv >> 1;       // 0..3 wave row
  const int wc   = wv & 1;        // 0..1 wave col
  const int lm   = lane & 15;
  const int lh   = lane >> 4;     // lane-half selector

  v8f acc[2][4];
#pragma unroll
  for (int i = 0; i < 2; ++i)
#pragma unroll
    for (int j = 0; j < 4; ++j) acc[i][j] = v8f_zero();

#ifdef CDNA5_HAS_TDM
  const unsigned long long gA = (unsigned long long)(size_t)(A  + (size_t)m0 * K);
  const unsigned long long gB = (unsigned long long)(size_t)(Bt + (size_t)n0 * K);
  const unsigned ldsA0 = (unsigned)(size_t)&As[0][0][0];
  const unsigned ldsA1 = (unsigned)(size_t)&As[1][0][0];
  const unsigned ldsB0 = (unsigned)(size_t)&Bs[0][0][0];
  const unsigned ldsB1 = (unsigned)(size_t)&Bs[1][0][0];
  if (wv == 0) {  // prologue: stage tile 0 via TDM
    tdm_load_2d(ldsA0, gA, K, M, K, 32, 128, 3, 3);
    tdm_load_2d(ldsB0, gB, K, N, K, 32, 128, 3, 3);
  }
  __builtin_amdgcn_s_wait_tensorcnt(0);
  __syncthreads();
#else
  const int srow = tid >> 1;             // 0..127 staging row
  const int sc0  = (tid & 1) * 16;       // 0 or 16 staging k-offset
  const bf16* agp = A  + (size_t)(m0 + srow) * K + sc0;
  const bf16* bgp = Bt + (size_t)(n0 + srow) * K + sc0;
  {
    uint4 a0 = *(const uint4*)(agp);
    uint4 a1 = *(const uint4*)(agp + 8);
    uint4 b0 = *(const uint4*)(bgp);
    uint4 b1 = *(const uint4*)(bgp + 8);
    *(uint4*)&As[0][srow][sc0]     = a0;
    *(uint4*)&As[0][srow][sc0 + 8] = a1;
    *(uint4*)&Bs[0][srow][sc0]     = b0;
    *(uint4*)&Bs[0][srow][sc0 + 8] = b1;
  }
#endif

  int cur = 0;
  for (int k0 = 0; k0 < K; k0 += 32) {
    const bool more = (k0 + 32) < K;
#ifdef CDNA5_HAS_TDM
    if (more && wv == 0) {  // stage next tile while computing current
      unsigned long long goff = (unsigned long long)(k0 + 32) * 2;
      tdm_load_2d(cur ? ldsA0 : ldsA1, gA + goff, K, M, K, 32, 128, 3, 3);
      tdm_load_2d(cur ? ldsB0 : ldsB1, gB + goff, K, N, K, 32, 128, 3, 3);
    }
#else
    __syncthreads();
    uint4 na0, na1, nb0, nb1;
    if (more) {
      const bf16* agn = agp + k0 + 32;
      const bf16* bgn = bgp + k0 + 32;
      na0 = *(const uint4*)(agn);
      na1 = *(const uint4*)(agn + 8);
      nb0 = *(const uint4*)(bgn);
      nb1 = *(const uint4*)(bgn + 8);
      if (k0 + 64 < K) {
        __builtin_prefetch(agn + 32, 0, 0);
        __builtin_prefetch(bgn + 32, 0, 0);
      }
    }
#endif

    // preload all fragments, then 8 back-to-back WMMAs
    Frag16 af[2], bfv[4];
    const int kb8  = lh * 8;
    const int kb16 = lh * 16;
#pragma unroll
    for (int tm = 0; tm < 2; ++tm) {
      int r = wr * 32 + tm * 16 + lm;
      af[tm].u[0] = *(const uint4*)&As[cur][r][kb8];       // K 0-7 / 8-15
      af[tm].u[1] = *(const uint4*)&As[cur][r][kb8 + 16];  // K 16-23 / 24-31
    }
#pragma unroll
    for (int tn = 0; tn < 4; ++tn) {
      int bcol = wc * 64 + tn * 16 + lm;
      bfv[tn].u[0] = *(const uint4*)&Bs[cur][bcol][kb8];
      bfv[tn].u[1] = *(const uint4*)&Bs[cur][bcol][kb8 + 16];
    }
#pragma unroll
    for (int tn = 0; tn < 4; ++tn) {
      acc[0][tn] = wmma_bf16(af[0].v, bfv[tn].v, acc[0][tn]);
      acc[1][tn] = wmma_bf16(af[1].v, bfv[tn].v, acc[1][tn]);
    }

#ifdef CDNA5_HAS_TDM
    __builtin_amdgcn_s_wait_tensorcnt(0);
    __syncthreads();
#else
    if (more) {
      const int nxt = cur ^ 1;
      *(uint4*)&As[nxt][srow][sc0]     = na0;
      *(uint4*)&As[nxt][srow][sc0 + 8] = na1;
      *(uint4*)&Bs[nxt][srow][sc0]     = nb0;
      *(uint4*)&Bs[nxt][srow][sc0 + 8] = nb1;
    }
#endif
    cur ^= 1;
  }

  // epilogue: bias (+ReLU), write f32 and/or bf16 (compile-time selected)
#pragma unroll
  for (int tm = 0; tm < 2; ++tm)
#pragma unroll
    for (int tn = 0; tn < 4; ++tn) {
      int col = n0 + wc * 64 + tn * 16 + lm;
      float bv = bias[col];
#pragma unroll
      for (int r = 0; r < 8; ++r) {
        int row = m0 + wr * 32 + tm * 16 + r + lh * 8;
        float v = acc[tm][tn][r] + bv;
        if constexpr (RELU) v = fmaxf(v, 0.0f);
        size_t idx = (size_t)row * N + col;
        if constexpr (HASF) outF[idx] = v;
        if constexpr (HASB) outB[idx] = (bf16)v;
      }
    }
}

// ---------------------------------------------------------------------------
// Flash-style attention. Grid: (B*H, S/64). Block: 128 (4 waves).
// Each wave owns a 16-row query strip; loop over 64-key blocks with running
// max/sum; QK^T and P*V via WMMA; softmax in f32 split across lane pairs.
// V comes pre-transposed as Vt[B,H,dh,S]. K/V tiles staged by TDM if present.
// ---------------------------------------------------------------------------
__global__ __launch_bounds__(128) void attn_kernel(
    const bf16* __restrict__ Q, const bf16* __restrict__ Kp,
    const bf16* __restrict__ Vt, const float* __restrict__ mask,
    bf16* __restrict__ ctx) {
  __shared__ bf16  Ks[64][72];        // [key][d]
  __shared__ bf16  Vs[64][72];        // [d][key]
  __shared__ float Sc[4][16][68];     // per-wave score strip (f32)
  __shared__ bf16  Ps[4][16][72];     // per-wave probability strip (bf16)
  __shared__ float mrun[4][16], lrun[4][16], corr[4][16];

  const int tid  = threadIdx.x;
  const int lane = tid & 31;
  const int w    = tid >> 5;
  const int lm   = lane & 15;
  const int lh   = lane >> 4;
  const int b    = blockIdx.x >> 4;
  const int hh   = blockIdx.x & 15;
  const int q0   = blockIdx.y * 64;
  const size_t headoff = (size_t)hh * DH;

  // Preload Q fragments for this wave's 16 rows (dh=64 -> 2 K-steps).
  Frag16 qa[2];
  {
    int qrow = q0 + w * 16 + lm;
    const bf16* qp = Q + ((size_t)(b * SEQ + qrow)) * DMODEL + headoff;
#pragma unroll
    for (int ks = 0; ks < 2; ++ks) {
      int kb = ks * 32 + lh * 8;
      qa[ks].u[0] = *(const uint4*)(qp + kb);
      qa[ks].u[1] = *(const uint4*)(qp + kb + 16);
    }
  }
  if (lane < 16) { mrun[w][lm] = -1e30f; lrun[w][lm] = 0.0f; }

  v8f cacc[4];
#pragma unroll
  for (int tn = 0; tn < 4; ++tn) cacc[tn] = v8f_zero();

  const bf16* vtbase = Vt + (size_t)(b * H_HEADS + hh) * DH * SEQ;
#ifdef CDNA5_HAS_TDM
  const unsigned ldsK = (unsigned)(size_t)&Ks[0][0];
  const unsigned ldsV = (unsigned)(size_t)&Vs[0][0];
#else
  const int krow = tid >> 1;         // 0..63
  const int kc0  = (tid & 1) * 32;   // 0 or 32
#endif

  for (int kb0 = 0; kb0 < SEQ; kb0 += 64) {
#ifdef CDNA5_HAS_TDM
    if (w == 0) {
      const bf16* kpt = Kp + ((size_t)(b * SEQ + kb0)) * DMODEL + headoff;
      tdm_load_2d(ldsK, (unsigned long long)(size_t)kpt,
                  DMODEL, SEQ, DMODEL, 64, 64, 4, 3);     // K tile [key][d]
      const bf16* vpt = vtbase + kb0;
      tdm_load_2d(ldsV, (unsigned long long)(size_t)vpt,
                  SEQ, DH, SEQ, 64, 64, 4, 3);            // V tile [d][key]
    }
    __builtin_amdgcn_s_wait_tensorcnt(0);
    __syncthreads();
#else
    const bf16* kp = Kp + ((size_t)(b * SEQ + kb0 + krow)) * DMODEL + headoff + kc0;
    *(uint4*)&Ks[krow][kc0]      = *(const uint4*)(kp);
    *(uint4*)&Ks[krow][kc0 + 8]  = *(const uint4*)(kp + 8);
    *(uint4*)&Ks[krow][kc0 + 16] = *(const uint4*)(kp + 16);
    *(uint4*)&Ks[krow][kc0 + 24] = *(const uint4*)(kp + 24);
    const bf16* vp = vtbase + (size_t)krow * SEQ + kb0 + kc0;
    *(uint4*)&Vs[krow][kc0]      = *(const uint4*)(vp);
    *(uint4*)&Vs[krow][kc0 + 8]  = *(const uint4*)(vp + 8);
    *(uint4*)&Vs[krow][kc0 + 16] = *(const uint4*)(vp + 16);
    *(uint4*)&Vs[krow][kc0 + 24] = *(const uint4*)(vp + 24);
    __syncthreads();
#endif

    // --- scores: S = Q * K^T * (1/sqrt(dh)) ---
#pragma unroll
    for (int tn = 0; tn < 4; ++tn) {
      v8f s = v8f_zero();
#pragma unroll
      for (int ks = 0; ks < 2; ++ks) {
        int keycol = tn * 16 + lm;
        int kbb = ks * 32 + lh * 8;
        Frag16 bfv;
        bfv.u[0] = *(const uint4*)&Ks[keycol][kbb];
        bfv.u[1] = *(const uint4*)&Ks[keycol][kbb + 16];
        s = wmma_bf16(qa[ks].v, bfv.v, s);
      }
#pragma unroll
      for (int r = 0; r < 8; ++r)
        Sc[w][r + lh * 8][tn * 16 + lm] = s[r] * 0.125f;
    }
    __syncthreads();

    // --- online softmax: row lm, lane pair (lh) splits 64 keys into 2x32 ---
    {
      const int r = lm;
      const int qrow = q0 + w * 16 + r;
      const int jbase = lh * 32;
      const float* mrow =
          mask + ((size_t)b * SEQ + qrow) * SEQ + kb0 + jbase;
      float sv[32];
      float mb = -1e30f;
#pragma unroll
      for (int q4 = 0; q4 < 8; ++q4) {
        float4 m4 = *(const float4*)(mrow + q4 * 4);
        float4 s4 = *(const float4*)&Sc[w][r][jbase + q4 * 4];
        sv[q4 * 4 + 0] = s4.x - 1e9f * m4.x;
        sv[q4 * 4 + 1] = s4.y - 1e9f * m4.y;
        sv[q4 * 4 + 2] = s4.z - 1e9f * m4.z;
        sv[q4 * 4 + 3] = s4.w - 1e9f * m4.w;
#pragma unroll
        for (int j = 0; j < 4; ++j) mb = fmaxf(mb, sv[q4 * 4 + j]);
      }
      mb = fmaxf(mb, __shfl_xor(mb, 16, 32));  // merge lane pair
      float mold = mrun[w][r];
      mb = fmaxf(mb, mold);
      float c = __expf(mold - mb);
      float lsum = 0.0f;
#pragma unroll
      for (int j = 0; j < 32; j += 2) {
        float p0 = __expf(sv[j] - mb);
        float p1 = __expf(sv[j + 1] - mb);
        lsum += p0 + p1;
        union { bf16 h[2]; unsigned u; } pk;
        pk.h[0] = (bf16)p0;
        pk.h[1] = (bf16)p1;
        *(unsigned*)&Ps[w][r][jbase + j] = pk.u;
      }
      lsum += __shfl_xor(lsum, 16, 32);
      if (lane < 16) {
        corr[w][r] = c;
        lrun[w][r] = lrun[w][r] * c + lsum;
        mrun[w][r] = mb;
      }
    }
    __syncthreads();

    // --- ctx = ctx * corr + P * V ---
    float cr[8];
#pragma unroll
    for (int r = 0; r < 8; ++r) cr[r] = corr[w][r + lh * 8];
    Frag16 pa[2];
#pragma unroll
    for (int ks = 0; ks < 2; ++ks) {
      int kbb = ks * 32 + lh * 8;
      pa[ks].u[0] = *(const uint4*)&Ps[w][lm][kbb];
      pa[ks].u[1] = *(const uint4*)&Ps[w][lm][kbb + 16];
    }
#pragma unroll
    for (int tn = 0; tn < 4; ++tn) {
#pragma unroll
      for (int r = 0; r < 8; ++r) cacc[tn][r] *= cr[r];
#pragma unroll
      for (int ks = 0; ks < 2; ++ks) {
        int dcol = tn * 16 + lm;
        int kbb = ks * 32 + lh * 8;
        Frag16 bfv;
        bfv.u[0] = *(const uint4*)&Vs[dcol][kbb];
        bfv.u[1] = *(const uint4*)&Vs[dcol][kbb + 16];
        cacc[tn] = wmma_bf16(pa[ks].v, bfv.v, cacc[tn]);
      }
    }
    __syncthreads();
  }

  // --- finalize: divide by running sum, store bf16 context ---
  float lr[8];
#pragma unroll
  for (int r = 0; r < 8; ++r) lr[r] = lrun[w][r + lh * 8];
#pragma unroll
  for (int tn = 0; tn < 4; ++tn)
#pragma unroll
    for (int r = 0; r < 8; ++r) {
      int srow = q0 + w * 16 + r + lh * 8;
      float v = cacc[tn][r] / lr[r];
      ctx[((size_t)(b * SEQ + srow)) * DMODEL + headoff + tn * 16 + lm] = (bf16)v;
    }
}

// ---------------------------------------------------------------------------
// out = LayerNorm(a + c): gamma*(x-mean)/(std+eps)+beta, population std.
// ---------------------------------------------------------------------------
__global__ __launch_bounds__(256) void resln_kernel(
    const float* __restrict__ a, const float* __restrict__ c,
    const float* __restrict__ g, const float* __restrict__ be,
    float* __restrict__ outF, bf16* __restrict__ outB) {
  __shared__ float red[256];
  const int row = blockIdx.x, t = threadIdx.x;
  const float* ar = a + (size_t)row * DMODEL;
  const float* cr = c + (size_t)row * DMODEL;
  float x[4];
  float s = 0.0f;
#pragma unroll
  for (int i = 0; i < 4; ++i) { x[i] = ar[t + i * 256] + cr[t + i * 256]; s += x[i]; }
  red[t] = s; __syncthreads();
  for (int off = 128; off > 0; off >>= 1) {
    if (t < off) red[t] += red[t + off];
    __syncthreads();
  }
  float mean = red[0] * (1.0f / DMODEL);
  __syncthreads();
  float s2 = 0.0f;
#pragma unroll
  for (int i = 0; i < 4; ++i) { float d = x[i] - mean; s2 += d * d; }
  red[t] = s2; __syncthreads();
  for (int off = 128; off > 0; off >>= 1) {
    if (t < off) red[t] += red[t + off];
    __syncthreads();
  }
  float var = red[0] * (1.0f / DMODEL);
  float inv = 1.0f / (sqrtf(var) + LN_EPS);
#pragma unroll
  for (int i = 0; i < 4; ++i) {
    int j = t + i * 256;
    float v = g[j] * (x[i] - mean) * inv + be[j];
    outF[(size_t)row * DMODEL + j] = v;
    outB[(size_t)row * DMODEL + j] = (bf16)v;
  }
}

// ---------------------------------------------------------------------------
extern "C" void kernel_launch(void* const* d_in, const int* in_sizes, int n_in,
                              void* d_out, int out_size, void* d_ws, size_t ws_size,
                              hipStream_t stream) {
  (void)in_sizes; (void)n_in; (void)out_size; (void)ws_size;
  const float* x    = (const float*)d_in[0];
  const float* mask = (const float*)d_in[1];
  const float* Wq   = (const float*)d_in[2];
  const float* bq   = (const float*)d_in[3];
  const float* Wk   = (const float*)d_in[4];
  const float* bk   = (const float*)d_in[5];
  const float* Wv   = (const float*)d_in[6];
  const float* bv   = (const float*)d_in[7];
  const float* Wo   = (const float*)d_in[8];
  const float* bo   = (const float*)d_in[9];
  const float* W1   = (const float*)d_in[10];
  const float* b1   = (const float*)d_in[11];
  const float* W2   = (const float*)d_in[12];
  const float* b2   = (const float*)d_in[13];
  const float* g1   = (const float*)d_in[14];
  const float* be1  = (const float*)d_in[15];
  const float* g2   = (const float*)d_in[16];
  const float* be2  = (const float*)d_in[17];
  float* out = (float*)d_out;

  char* ws = (char*)d_ws;
  size_t off = 0;
  auto carve = [&](size_t bytes) -> char* {
    char* p = ws + off;
    off = (off + bytes + 255) & ~(size_t)255;
    return p;
  };
  const size_t NE = (size_t)NROWS * DMODEL;  // 4M elems
  bf16*  hb   = (bf16*)carve(NE * 2);
  float* hf   = (float*)carve(NE * 4);
  bf16*  qb   = (bf16*)carve(NE * 2);
  bf16*  kb   = (bf16*)carve(NE * 2);
  bf16*  vb   = (bf16*)carve(NE * 2);
  bf16*  vbT  = (bf16*)carve(NE * 2);
  bf16*  ctxb = (bf16*)carve(NE * 2);
  float* attf = (float*)carve(NE * 4);
  float* o1f  = (float*)carve(NE * 4);
  bf16*  o1b  = (bf16*)carve(NE * 2);
  bf16*  f1b  = (bf16*)carve((size_t)NROWS * MUDIM * 2);
  float* f2f  = (float*)carve(NE * 4);
  // per-layer bf16 transposed weights (reused each layer)
  bf16* wqT = (bf16*)carve((size_t)DMODEL * DMODEL * 2);
  bf16* wkT = (bf16*)carve((size_t)DMODEL * DMODEL * 2);
  bf16* wvT = (bf16*)carve((size_t)DMODEL * DMODEL * 2);
  bf16* woT = (bf16*)carve((size_t)DMODEL * DMODEL * 2);
  bf16* w1T = (bf16*)carve((size_t)DMODEL * MUDIM * 2);
  bf16* w2T = (bf16*)carve((size_t)MUDIM * DMODEL * 2);

  concat_kernel<<<NROWS, 256, 0, stream>>>(x, hf, hb);

  dim3 gD (NROWS / 128, DMODEL / 128);        // (32, 8)
  dim3 gMU(NROWS / 128, MUDIM / 128);         // (32, 16)
  dim3 gAt(BATCH * H_HEADS, SEQ / 64);        // (128, 8)
  dim3 gWd(DMODEL / 32, DMODEL / 32);         // (32, 32)
  dim3 gW1(DMODEL / 32, MUDIM / 32);          // (32, 64)
  dim3 gW2(MUDIM / 32, DMODEL / 32);          // (64, 32)
  dim3 gVt(BATCH * H_HEADS, SEQ / 32, DH / 32);

  for (int i = 0; i < L_LAYERS; ++i) {
    const float* wq = Wq + (size_t)i * DMODEL * DMODEL;
    const float* wk = Wk + (size_t)i * DMODEL * DMODEL;
    const float* wv = Wv + (size_t)i * DMODEL * DMODEL;
    const float* wo = Wo + (size_t)i * DMODEL * DMODEL;
    const float* w1 = W1 + (size_t)i * DMODEL * MUDIM;
    const float* w2 = W2 + (size_t)i * MUDIM * DMODEL;

    wtrans_kernel<<<gWd, 256, 0, stream>>>(wq, wqT, DMODEL, DMODEL);
    wtrans_kernel<<<gWd, 256, 0, stream>>>(wk, wkT, DMODEL, DMODEL);
    wtrans_kernel<<<gWd, 256, 0, stream>>>(wv, wvT, DMODEL, DMODEL);
    wtrans_kernel<<<gWd, 256, 0, stream>>>(wo, woT, DMODEL, DMODEL);
    wtrans_kernel<<<gW1, 256, 0, stream>>>(w1, w1T, DMODEL, MUDIM);
    wtrans_kernel<<<gW2, 256, 0, stream>>>(w2, w2T, MUDIM, DMODEL);

    gemm_kernel<false, true, false><<<gD, 256, 0, stream>>>(
        hb, wqT, bq + i * DMODEL, nullptr, qb, NROWS, DMODEL, DMODEL);
    gemm_kernel<false, true, false><<<gD, 256, 0, stream>>>(
        hb, wkT, bk + i * DMODEL, nullptr, kb, NROWS, DMODEL, DMODEL);
    gemm_kernel<false, true, false><<<gD, 256, 0, stream>>>(
        hb, wvT, bv + i * DMODEL, nullptr, vb, NROWS, DMODEL, DMODEL);
    vtrans_kernel<<<gVt, 256, 0, stream>>>(vb, vbT);
    attn_kernel<<<gAt, 128, 0, stream>>>(qb, kb, vbT, mask, ctxb);
    gemm_kernel<true, false, false><<<gD, 256, 0, stream>>>(
        ctxb, woT, bo + i * DMODEL, attf, nullptr, NROWS, DMODEL, DMODEL);
    resln_kernel<<<NROWS, 256, 0, stream>>>(hf, attf, g1 + i * DMODEL,
                                            be1 + i * DMODEL, o1f, o1b);
    gemm_kernel<false, true, true><<<gMU, 256, 0, stream>>>(
        o1b, w1T, b1 + i * MUDIM, nullptr, f1b, NROWS, MUDIM, DMODEL);
    gemm_kernel<true, false, true><<<gD, 256, 0, stream>>>(
        f1b, w2T, b2 + i * DMODEL, f2f, nullptr, NROWS, DMODEL, MUDIM);
    float* dstF = (i == L_LAYERS - 1) ? out : hf;
    resln_kernel<<<NROWS, 256, 0, stream>>>(o1f, f2f, g2 + i * DMODEL,
                                            be2 + i * DMODEL, dstF, hb);
  }
}
